// SlotValueGenerator_66262755443201
// MI455X (gfx1250) — matrix operand order repaired and checked by reading, not compiled, not run on hardware
//
#include <hip/hip_runtime.h>

#define SVG_D 768   // hidden dim (fixed by reference)

typedef __bf16 bf16;
typedef __attribute__((ext_vector_type(16))) __bf16 v16bf;
typedef __attribute__((ext_vector_type(8)))  __bf16 v8bf;
typedef __attribute__((ext_vector_type(8)))  float  v8f;

__device__ __forceinline__ bf16 f32_to_bf16(float f) {
  unsigned u = __float_as_uint(f);
  unsigned r = u + 0x7FFFu + ((u >> 16) & 1u);   // round-to-nearest-even
  unsigned short hs = (unsigned short)(r >> 16);
  bf16 out;
  __builtin_memcpy(&out, &hs, sizeof(out));
  return out;
}

// ---------------- one-time: emb fp32 -> bf16 (rows >= V zero-padded) -------
__global__ void svg_conv_emb(const float* __restrict__ emb, bf16* __restrict__ emb_bf,
                             int V, int Vpad) {
  size_t i = (size_t)blockIdx.x * blockDim.x + threadIdx.x;
  size_t total = (size_t)Vpad * SVG_D;
  if (i >= total) return;
  int v = (int)(i / SVG_D);
  float f = 0.0f;
  if (v < V) f = emb[i];
  emb_bf[i] = f32_to_bf16(f);
}

__global__ void svg_copy_f32(const float* __restrict__ src, float* __restrict__ dst, int n) {
  int i = blockIdx.x * blockDim.x + threadIdx.x;
  if (i < n) dst[i] = src[i];
}

// x[b,d] = dec[b, j, d]  (dec is (B,J,D))
__global__ void svg_init_x(const float* __restrict__ dec, float* __restrict__ x,
                           int J, int j, int n) {
  int i = blockIdx.x * blockDim.x + threadIdx.x;
  if (i >= n) return;
  int b = i / SVG_D, d = i - b * SVG_D;
  x[i] = dec[((size_t)b * J + j) * SVG_D + d];
}

// per-step: zero p_ctx (B*Vpad floats) and argmax keys (B u64)
__global__ void svg_zero_step(float* __restrict__ p_ctx, unsigned long long* __restrict__ amax,
                              int n, int B) {
  int i = blockIdx.x * blockDim.x + threadIdx.x;
  if (i < n) p_ctx[i] = 0.0f;
  if (i < B) amax[i] = 0ull;
}

// ---------------- GRU cell (fp32, keeps the recurrence in full precision) --
__global__ __launch_bounds__(256) void svg_gru(
    const float* __restrict__ x, const float* __restrict__ h,
    const float* __restrict__ w_ih, const float* __restrict__ w_hh,
    const float* __restrict__ b_ih, const float* __restrict__ b_hh,
    float* __restrict__ h_new, bf16* __restrict__ h_bf) {
  __shared__ float shx[SVG_D];
  __shared__ float shh[SVG_D];
  const int b = blockIdx.x;
  const int d = blockIdx.y * 256 + threadIdx.x;   // gridDim.y = D/256 -> d < D
  for (int i = threadIdx.x; i < SVG_D; i += 256) {
    shx[i] = x[(size_t)b * SVG_D + i];
    shh[i] = h[(size_t)b * SVG_D + i];
  }
  __syncthreads();
  const float* wi_r = w_ih + (size_t)d * SVG_D;
  const float* wi_z = w_ih + (size_t)(SVG_D + d) * SVG_D;
  const float* wi_n = w_ih + (size_t)(2 * SVG_D + d) * SVG_D;
  const float* wh_r = w_hh + (size_t)d * SVG_D;
  const float* wh_z = w_hh + (size_t)(SVG_D + d) * SVG_D;
  const float* wh_n = w_hh + (size_t)(2 * SVG_D + d) * SVG_D;
  float ir = b_ih[d], iz = b_ih[SVG_D + d], in_ = b_ih[2 * SVG_D + d];
  float hr = b_hh[d], hz = b_hh[SVG_D + d], hn = b_hh[2 * SVG_D + d];
  for (int k = 0; k < SVG_D; ++k) {
    float xv = shx[k], hv = shh[k];
    ir = fmaf(xv, wi_r[k], ir);  iz = fmaf(xv, wi_z[k], iz);  in_ = fmaf(xv, wi_n[k], in_);
    hr = fmaf(hv, wh_r[k], hr);  hz = fmaf(hv, wh_z[k], hz);  hn  = fmaf(hv, wh_n[k], hn);
  }
  float r = 1.0f / (1.0f + __expf(-(ir + hr)));
  float z = 1.0f / (1.0f + __expf(-(iz + hz)));
  float n = tanhf(in_ + r * hn);
  float out = (1.0f - z) * n + z * shh[d];
  h_new[(size_t)b * SVG_D + d] = out;
  h_bf[(size_t)b * SVG_D + d] = f32_to_bf16(out);
}

// ------- attention + context + pointer scatter + p_gen gate (1 block / b) --
__global__ __launch_bounds__(256) void svg_attn(
    const float* __restrict__ enc, const int* __restrict__ ids,
    const float* __restrict__ h, const float* __restrict__ x,
    const float* __restrict__ w_gen, const float* __restrict__ b_gen,
    float* __restrict__ p_ctx, float* __restrict__ p_gen,
    int X, int Vpad) {
  __shared__ float shh[SVG_D];
  __shared__ float shc[SVG_D];
  __shared__ float attn[256];
  __shared__ float red[256];
  const int b = blockIdx.x, tid = threadIdx.x;
  for (int i = tid; i < SVG_D; i += 256) shh[i] = h[(size_t)b * SVG_D + i];
  __syncthreads();
  float s = -1e30f;
  int id = 0;
  if (tid < X) {
    const float* e = enc + ((size_t)b * X + tid) * SVG_D;
    float acc = 0.0f;
    for (int k = 0; k < SVG_D; ++k) acc = fmaf(e[k], shh[k], acc);
    id = ids[b * X + tid];
    s = (id == 0) ? -1e9f : acc;   // PAD mask
  }
  red[tid] = s; __syncthreads();
  for (int off = 128; off > 0; off >>= 1) {
    if (tid < off) red[tid] = fmaxf(red[tid], red[tid + off]);
    __syncthreads();
  }
  float m = red[0]; __syncthreads();
  float e = (tid < X) ? __expf(s - m) : 0.0f;
  red[tid] = e; __syncthreads();
  for (int off = 128; off > 0; off >>= 1) {
    if (tid < off) red[tid] += red[tid + off];
    __syncthreads();
  }
  float inv = 1.0f / red[0]; __syncthreads();
  attn[tid] = e * inv;
  __syncthreads();
  // deterministic scatter of copy probabilities (duplicate ids possible)
  if (tid == 0) {
    for (int xx = 0; xx < X; ++xx)
      p_ctx[(size_t)b * Vpad + ids[b * X + xx]] += attn[xx];
  }
  // context vector
  for (int d0 = tid; d0 < SVG_D; d0 += 256) {
    float c = 0.0f;
    for (int xx = 0; xx < X; ++xx)
      c = fmaf(attn[xx], enc[((size_t)b * X + xx) * SVG_D + d0], c);
    shc[d0] = c;
  }
  __syncthreads();
  // p_gen = sigmoid(w_gen . [h, x, context] + b_gen)
  float pg = 0.0f;
  for (int i = tid; i < 3 * SVG_D; i += 256) {
    float v = (i < SVG_D) ? shh[i]
            : (i < 2 * SVG_D) ? x[(size_t)b * SVG_D + (i - SVG_D)]
            : shc[i - 2 * SVG_D];
    pg = fmaf(w_gen[i], v, pg);
  }
  red[tid] = pg; __syncthreads();
  for (int off = 128; off > 0; off >>= 1) {
    if (tid < off) red[tid] += red[tid + off];
    __syncthreads();
  }
  if (tid == 0) p_gen[b] = 1.0f / (1.0f + __expf(-(red[0] + b_gen[0])));
}

// ---------------- vocab logits: (32 x 768) x (768 x Vpad) via bf16 WMMA ----
// Block: 256 threads = 8 waves; LDS holds h (32x768 bf16, zero-padded rows).
// Each wave owns ONE 16-column N-tile and computes BOTH 16-row M-tiles,
// so every global B-fragment (emb) is loaded once and used twice — halves
// the dominant emb/L2 read traffic vs a 1-tile-per-wave mapping.
__global__ __launch_bounds__(256) void svg_vocab_wmma(
    const bf16* __restrict__ emb_bf, const bf16* __restrict__ h_bf,
    float* __restrict__ logits, int B, int Vpad) {
  __shared__ bf16 sh[32 * SVG_D];   // 48 KB
  const int tid = threadIdx.x;
  for (int i = tid; i < 32 * SVG_D; i += 256) {
    int r = i / SVG_D;
    sh[i] = (r < B) ? h_bf[i] : f32_to_bf16(0.0f);
  }
  __syncthreads();
  const int w = tid >> 5, lane = tid & 31;
  const int hb = lane >> 4, ln = lane & 15;
  const int v0 = blockIdx.x * 128 + w * 16;   // 8 N-tiles per block
  // B-fragment: lane (hb,ln) holds emb[v0+ln, k0 + hb*16 .. +15] (contiguous!)
  const bf16* brow = emb_bf + (size_t)(v0 + ln) * SVG_D + hb * 16;
  // A-fragments: lane holds h[m, k0+hb*16..+15] (contiguous, matches B)
  const bf16* arow0 = sh + (ln)      * SVG_D + hb * 16;   // M-tile 0: rows 0..15
  const bf16* arow1 = sh + (16 + ln) * SVG_D + hb * 16;   // M-tile 1: rows 16..31
  v8f c0 = {};
  v8f c1 = {};
  for (int k0 = 0; k0 < SVG_D; k0 += 32) {
    if (k0 + 32 < SVG_D) __builtin_prefetch(brow + k0 + 32, 0, 0);
    v16bf bm = *(const v16bf*)(brow + k0);
    v8bf a0lo = *(const v8bf*)(arow0 + k0);
    v8bf a0hi = *(const v8bf*)(arow0 + k0 + 8);
    v8bf a1lo = *(const v8bf*)(arow1 + k0);
    v8bf a1hi = *(const v8bf*)(arow1 + k0 + 8);
    v16bf a0 = __builtin_shufflevector(a0lo, a0hi,
        0, 1, 2, 3, 4, 5, 6, 7, 8, 9, 10, 11, 12, 13, 14, 15);
    v16bf a1 = __builtin_shufflevector(a1lo, a1hi,
        0, 1, 2, 3, 4, 5, 6, 7, 8, 9, 10, 11, 12, 13, 14, 15);
    c0 = __builtin_amdgcn_wmma_f32_16x16x32_bf16(
        false, a0, false, bm, (short)0, c0, false, false);
    c1 = __builtin_amdgcn_wmma_f32_16x16x32_bf16(
        false, a1, false, bm, (short)0, c1, false, false);
  }
  // C/D layout: VGPR r -> M=r (lanes 0-15), M=r+8 (lanes 16-31); N = lane%16
  const int col = v0 + ln;
  const int mbase = hb * 8;
#pragma unroll
  for (int r = 0; r < 8; ++r) {
    int m0 = mbase + r;
    if (m0 < B) logits[(size_t)m0 * Vpad + col] = c0[r];
    int m1 = 16 + mbase + r;
    if (m1 < B) logits[(size_t)m1 * Vpad + col] = c1[r];
  }
}

// ---------------- row softmax stats (max, sum-exp), one block per row ------
__global__ __launch_bounds__(256) void svg_softmax_stats(
    const float* __restrict__ logits, float* __restrict__ rowmax,
    float* __restrict__ rowsum, int V, int Vpad) {
  __shared__ float red[256];
  const int b = blockIdx.x, tid = threadIdx.x;
  const float* row = logits + (size_t)b * Vpad;
  float m = -1e30f;
  for (int v = tid; v < V; v += 256) m = fmaxf(m, row[v]);
  red[tid] = m; __syncthreads();
  for (int off = 128; off > 0; off >>= 1) {
    if (tid < off) red[tid] = fmaxf(red[tid], red[tid + off]);
    __syncthreads();
  }
  m = red[0]; __syncthreads();
  float s = 0.0f;
  for (int v = tid; v < V; v += 256) s += __expf(row[v] - m);
  red[tid] = s; __syncthreads();
  for (int off = 128; off > 0; off >>= 1) {
    if (tid < off) red[tid] += red[tid + off];
    __syncthreads();
  }
  if (tid == 0) { rowmax[b] = m; rowsum[b] = red[0]; }
}

// ------- p_svg = pg*softmax + (1-pg)*p_ctx ; write out ; packed argmax -----
__global__ __launch_bounds__(256) void svg_out(
    const float* __restrict__ logits, const float* __restrict__ p_ctx,
    const float* __restrict__ p_gen, const float* __restrict__ rowmax,
    const float* __restrict__ rowsum, float* __restrict__ outb,
    size_t bstride, unsigned long long* __restrict__ amax, int V, int Vpad) {
  __shared__ unsigned long long red[256];
  const int b = blockIdx.x;
  const int v = blockIdx.y * 256 + threadIdx.x;
  unsigned long long key = 0ull;
  if (v < V) {
    float pg = p_gen[b];
    float pv = __expf(logits[(size_t)b * Vpad + v] - rowmax[b]) / rowsum[b];
    float p = pg * pv + (1.0f - pg) * p_ctx[(size_t)b * Vpad + v];
    outb[(size_t)b * bstride + v] = p;
    // p >= 0 -> float bits monotonic; secondary key prefers smaller v on ties
    key = ((unsigned long long)__float_as_uint(p) << 32) |
          (unsigned long long)(0xFFFFFFFFu - (unsigned)v);
  }
  red[threadIdx.x] = key; __syncthreads();
  for (int off = 128; off > 0; off >>= 1) {
    if (threadIdx.x < off) {
      unsigned long long o = red[threadIdx.x + off];
      if (o > red[threadIdx.x]) red[threadIdx.x] = o;
    }
    __syncthreads();
  }
  if (threadIdx.x == 0) atomicMax(&amax[b], red[0]);
}

// ---------------- greedy next input: x = emb[argmax] -----------------------
__global__ __launch_bounds__(256) void svg_nextx(
    const unsigned long long* __restrict__ amax, const float* __restrict__ emb,
    float* __restrict__ x) {
  const int b = blockIdx.x;
  unsigned v = 0xFFFFFFFFu - (unsigned)(amax[b] & 0xFFFFFFFFull);
  for (int d = threadIdx.x; d < SVG_D; d += 256)
    x[(size_t)b * SVG_D + d] = emb[(size_t)v * SVG_D + d];
}

// ===========================================================================
extern "C" void kernel_launch(void* const* d_in, const int* in_sizes, int n_in,
                              void* d_out, int out_size, void* d_ws, size_t ws_size,
                              hipStream_t stream) {
  const float* dec    = (const float*)d_in[0];
  const float* hidden = (const float*)d_in[1];
  const float* enc    = (const float*)d_in[2];
  const int*   ids    = (const int*)d_in[3];
  /* d_in[4] = max_len (device scalar; K derived from out_size instead) */
  const float* emb    = (const float*)d_in[5];
  const float* w_ih   = (const float*)d_in[6];
  const float* w_hh   = (const float*)d_in[7];
  const float* b_ih   = (const float*)d_in[8];
  const float* b_hh   = (const float*)d_in[9];
  const float* w_gen  = (const float*)d_in[10];
  const float* b_gen  = (const float*)d_in[11];

  const int B = in_sizes[1] / SVG_D;                 // 32
  const int J = in_sizes[0] / (B * SVG_D);           // 4
  const int X = in_sizes[2] / (B * SVG_D);           // 256
  const int V = in_sizes[5] / SVG_D;                 // 30522
  const int K = out_size / (B * J * V);              // 8
  const int Vpad = ((V + 127) / 128) * 128;          // 30592 (128 cols / block)

  // ---- workspace layout (256-B aligned slices) ----
  char* ws = (char*)d_ws;
  size_t off = 0;
  auto take = [&](size_t bytes) -> char* {
    char* p = ws + off;
    off = (off + bytes + 255) & ~(size_t)255;
    return p;
  };
  bf16*  emb_bf  = (bf16*)take((size_t)Vpad * SVG_D * sizeof(bf16));   // ~47 MB
  float* logits  = (float*)take((size_t)B * Vpad * sizeof(float));     // ~3.9 MB
  float* p_ctx   = (float*)take((size_t)B * Vpad * sizeof(float));     // ~3.9 MB
  float* hbuf0   = (float*)take((size_t)B * SVG_D * sizeof(float));
  float* hbuf1   = (float*)take((size_t)B * SVG_D * sizeof(float));
  bf16*  h_bf    = (bf16*)take((size_t)B * SVG_D * sizeof(bf16));
  float* x_f32   = (float*)take((size_t)B * SVG_D * sizeof(float));
  float* rowmax  = (float*)take((size_t)B * sizeof(float));
  float* rowsum  = (float*)take((size_t)B * sizeof(float));
  float* pg_buf  = (float*)take((size_t)B * sizeof(float));
  unsigned long long* amax =
      (unsigned long long*)take((size_t)B * sizeof(unsigned long long));
  (void)ws_size; (void)n_in;

  float* hbuf[2] = {hbuf0, hbuf1};

  // one-time: emb -> bf16 (L2-resident working set: ~47 MB of 192 MB L2)
  {
    size_t total = (size_t)Vpad * SVG_D;
    svg_conv_emb<<<(unsigned)((total + 255) / 256), 256, 0, stream>>>(emb, emb_bf, V, Vpad);
  }
  // h0 = hidden[0]
  svg_copy_f32<<<(B * SVG_D + 255) / 256, 256, 0, stream>>>(hidden, hbuf[0], B * SVG_D);

  int sc = 0;  // global step counter (h carries across j, ping-pong buffers)
  for (int j = 0; j < J; ++j) {
    svg_init_x<<<(B * SVG_D + 255) / 256, 256, 0, stream>>>(dec, x_f32, J, j, B * SVG_D);
    for (int k = 0; k < K; ++k) {
      float* hcur = hbuf[sc & 1];
      float* hnxt = hbuf[(sc + 1) & 1];
      ++sc;
      svg_zero_step<<<(unsigned)(((size_t)B * Vpad + 255) / 256), 256, 0, stream>>>(
          p_ctx, amax, B * Vpad, B);
      svg_gru<<<dim3(B, SVG_D / 256), 256, 0, stream>>>(
          x_f32, hcur, w_ih, w_hh, b_ih, b_hh, hnxt, h_bf);
      svg_attn<<<B, 256, 0, stream>>>(
          enc, ids, hnxt, x_f32, w_gen, b_gen, p_ctx, pg_buf, X, Vpad);
      svg_vocab_wmma<<<Vpad / 128, 256, 0, stream>>>(emb_bf, h_bf, logits, B, Vpad);
      svg_softmax_stats<<<B, 256, 0, stream>>>(logits, rowmax, rowsum, V, Vpad);
      svg_out<<<dim3(B, (V + 255) / 256), 256, 0, stream>>>(
          logits, p_ctx, pg_buf, rowmax, rowsum,
          (float*)d_out + ((size_t)j * K + k) * V, (size_t)J * K * V, amax, V, Vpad);
      svg_nextx<<<B, 256, 0, stream>>>(amax, emb, x_f32);
    }
  }
}
